// VarAttention_5669356834910
// MI455X (gfx1250) — matrix operand order repaired and checked by reading, not compiled, not run on hardware
//
#include <hip/hip_runtime.h>

// ---------------------------------------------------------------------------
// Types for CDNA5 WMMA (wave32): D(16x16 f32) = A(16x32 bf16) x B(32x16 bf16) + C
// ---------------------------------------------------------------------------
typedef __attribute__((ext_vector_type(16))) __bf16 bf16x16;
typedef __attribute__((ext_vector_type(8)))  float  floatx8;
typedef __attribute__((ext_vector_type(8)))  unsigned short ushort8;

union Frag {
    bf16x16 v;
    ushort8 h[2];
};

__device__ __forceinline__ unsigned short f2bf(float f) {
    union { float f; unsigned u; } x; x.f = f;
    unsigned r = x.u + 0x7FFFu + ((x.u >> 16) & 1u);   // round-to-nearest-even
    return (unsigned short)(r >> 16);
}
__device__ __forceinline__ float bf2f(unsigned short h) {
    union { unsigned u; float f; } x; x.u = ((unsigned)h) << 16;
    return x.f;
}
__device__ __forceinline__ floatx8 wmma_bf16(const Frag& a, const Frag& b, floatx8 c) {
    return __builtin_amdgcn_wmma_f32_16x16x32_bf16(false, a.v, false, b.v,
                                                   (short)0, c, false, false);
}

// gfx1250 async global->LDS copy (ASYNCcnt path). The builtin takes
// v4i pointers: global (AS1) source, LDS (AS3) destination.
#if __has_builtin(__builtin_amdgcn_global_load_async_to_lds_b128) && \
    __has_builtin(__builtin_amdgcn_s_wait_asynccnt)
#define HAVE_ASYNC_LDS 1
typedef int v4i __attribute__((vector_size(16)));
typedef __attribute__((address_space(1))) v4i* as1_v4i_ptr;
typedef __attribute__((address_space(3))) v4i* as3_v4i_ptr;
__device__ __forceinline__ void async_ld_b128(const void* g, void* l) {
    __builtin_amdgcn_global_load_async_to_lds_b128(
        (as1_v4i_ptr)g, (as3_v4i_ptr)l, 0, 0);
}
#else
#define HAVE_ASYNC_LDS 0
#endif

#define BATCH  16
#define NTOK   1569
#define DMODEL 768
#define NHEAD  12
#define DHEAD  64
#define MROWS  (BATCH * NTOK)   // 25104
#define NFR    196              // tokens per frame
#define L2E    1.4426950408889634f

// ---------------------------------------------------------------------------
// Kernel 1: qkv = x @ w_qkv  (fp32 in -> bf16 head-major q(scaled)/k/v out)
// Block tile 128(M) x 64(N), K-step 64. 8 waves = 4x2 of 32x32 wave tiles,
// each wave runs 8 WMMAs per staging round (2 K-chunks of 32).
// ---------------------------------------------------------------------------
__global__ __launch_bounds__(256) void gemm_qkv(
    const float* __restrict__ x, const float* __restrict__ w,
    unsigned short* __restrict__ qws, unsigned short* __restrict__ kws,
    unsigned short* __restrict__ vws)
{
    __shared__ __align__(16) unsigned short sA[128 * 72]; // [m][k], stride 72 (144B)
    __shared__ __align__(16) unsigned short sB[64 * 72];  // [n][k] transposed

    const int m0   = blockIdx.x * 128;
    const int n0   = blockIdx.y * 64;
    const int tid  = threadIdx.x;
    const int lane = tid & 31;
    const int wv   = tid >> 5;
    const int mw   = (wv & 3) * 32;
    const int nw   = (wv >> 2) * 32;
    const int arow = lane & 15;
    const int kb   = (lane >> 4) * 8;   // unused; A/B frags use kh16
    const int kh16 = (lane >> 4) * 16;  // B-frag K base (0 or 16)

    floatx8 acc[2][2] = {};

    for (int k0 = 0; k0 < DMODEL; k0 += 64) {
        // A tile: 128x64 fp32 -> bf16 LDS (2048 float4 chunks, 8 per thread)
        for (int id = tid; id < 2048; id += 256) {
            int row = id >> 4, c4 = (id & 15) << 2;
            int gm = m0 + row;
            float4 f = make_float4(0.f, 0.f, 0.f, 0.f);
            if (gm < MROWS) f = *(const float4*)(x + (size_t)gm * DMODEL + k0 + c4);
            unsigned lo = (unsigned)f2bf(f.x) | ((unsigned)f2bf(f.y) << 16);
            unsigned hi = (unsigned)f2bf(f.z) | ((unsigned)f2bf(f.w) << 16);
            *(uint2*)(&sA[row * 72 + c4]) = make_uint2(lo, hi);
        }
        // B tile: 64x64 fp32, stored transposed [n][k]
        for (int id = tid; id < 1024; id += 256) {
            int kr = id >> 4, n4 = (id & 15) << 2;
            float4 f = *(const float4*)(w + (size_t)(k0 + kr) * (3 * DMODEL) + n0 + n4);
            sB[(n4 + 0) * 72 + kr] = f2bf(f.x);
            sB[(n4 + 1) * 72 + kr] = f2bf(f.y);
            sB[(n4 + 2) * 72 + kr] = f2bf(f.z);
            sB[(n4 + 3) * 72 + kr] = f2bf(f.w);
        }
        // prefetch next K tiles into cache (global_prefetch_b8)
        if (k0 + 64 < DMODEL) {
            int gm = m0 + (tid >> 1);
            if (gm < MROWS)
                __builtin_prefetch(x + (size_t)gm * DMODEL + k0 + 64 + (tid & 1) * 32, 0, 0);
            __builtin_prefetch(w + (size_t)(k0 + 64 + (tid & 63)) * (3 * DMODEL)
                                 + n0 + (tid >> 6) * 16, 0, 0);
        }
        __syncthreads();

        for (int kc = 0; kc < 2; ++kc) {
            Frag af[2], bf_[2];
            for (int mf = 0; mf < 2; ++mf) {
                const unsigned short* p = &sA[(mw + mf * 16 + arow) * 72 + kc * 32];
                af[mf].h[0] = *(const ushort8*)(p + kh16);
                af[mf].h[1] = *(const ushort8*)(p + kh16 + 8);
            }
            for (int nf = 0; nf < 2; ++nf) {
                const unsigned short* p = &sB[(nw + nf * 16 + (lane & 15)) * 72 + kc * 32 + kh16];
                bf_[nf].h[0] = *(const ushort8*)(p);
                bf_[nf].h[1] = *(const ushort8*)(p + 8);
            }
            for (int mf = 0; mf < 2; ++mf)
                for (int nf = 0; nf < 2; ++nf)
                    acc[mf][nf] = wmma_bf16(af[mf], bf_[nf], acc[mf][nf]);
        }
        __syncthreads();
    }

    // Epilogue: scatter to head-major bf16 q/k/v; scale q by d^-0.5 = 0.125
    for (int mf = 0; mf < 2; ++mf)
        for (int nf = 0; nf < 2; ++nf) {
            int gcol   = n0 + nw + nf * 16 + (lane & 15);   // [0, 2304)
            int which  = gcol / DMODEL;
            int within = gcol - which * DMODEL;
            int head = within >> 6, dc = within & 63;
            unsigned short* dst = (which == 0) ? qws : (which == 1) ? kws : vws;
            float scale = (which == 0) ? 0.125f : 1.0f;
            for (int i = 0; i < 8; ++i) {
                int gm = m0 + mw + mf * 16 + (lane >> 4) * 8 + i;
                if (gm < MROWS) {
                    int b = gm / NTOK;
                    int n = gm - b * NTOK;
                    dst[(((size_t)b * NHEAD + head) * NTOK + n) * DHEAD + dc] =
                        f2bf(acc[mf][nf][i] * scale);
                }
            }
        }
}

// ---------------------------------------------------------------------------
// Kernel 2: cls-token attention: row 0 of each (b,h) attends over all 1569 keys
// ---------------------------------------------------------------------------
__global__ __launch_bounds__(256) void cls_attn(
    const unsigned short* __restrict__ qws, const unsigned short* __restrict__ kws,
    const unsigned short* __restrict__ vws, unsigned short* __restrict__ ows)
{
    __shared__ float sQ[64];
    __shared__ float sSim[NTOK];
    __shared__ float sRed[256];
    __shared__ float sPart[4][64];

    const int bh = blockIdx.x;
    const size_t base = (size_t)bh * NTOK * DHEAD;
    const int tid = threadIdx.x;

    if (tid < 64) sQ[tid] = bf2f(qws[base + tid]);  // q already scaled
    __syncthreads();

    float lmax = -1e30f;
    for (int j = tid; j < NTOK; j += 256) {
        const unsigned short* kr = kws + base + (size_t)j * DHEAD;
        float s = 0.f;
        for (int c = 0; c < 64; c += 2) {
            unsigned u = *(const unsigned*)(kr + c);
            s += sQ[c]     * bf2f((unsigned short)(u & 0xFFFFu));
            s += sQ[c + 1] * bf2f((unsigned short)(u >> 16));
        }
        sSim[j] = s;
        lmax = fmaxf(lmax, s);
    }
    sRed[tid] = lmax; __syncthreads();
    for (int s = 128; s > 0; s >>= 1) {
        if (tid < s) sRed[tid] = fmaxf(sRed[tid], sRed[tid + s]);
        __syncthreads();
    }
    float m = sRed[0]; __syncthreads();

    float lsum = 0.f;
    for (int j = tid; j < NTOK; j += 256) {
        float p = exp2f((sSim[j] - m) * L2E);
        sSim[j] = p;
        lsum += p;
    }
    __syncthreads();
    sRed[tid] = lsum; __syncthreads();
    for (int s = 128; s > 0; s >>= 1) {
        if (tid < s) sRed[tid] += sRed[tid + s];
        __syncthreads();
    }
    float l = sRed[0];

    int d = tid & 63, part = tid >> 6;
    float acc = 0.f;
    for (int j = part; j < NTOK; j += 4)
        acc += sSim[j] * bf2f(vws[base + (size_t)j * DHEAD + d]);
    sPart[part][d] = acc; __syncthreads();
    if (tid < 64) {
        float o = (sPart[0][tid] + sPart[1][tid] + sPart[2][tid] + sPart[3][tid]) / l;
        ows[base + tid] = f2bf(o);
    }
}

// ---------------------------------------------------------------------------
// Kernel 3: frame attention. One block per (bh, frame): 196 queries x 197 keys.
// Flash-style online softmax; QK^T and PV via WMMA bf16. K-tile staged with
// GLOBAL_LOAD_ASYNC_TO_LDS_B128 when available (ASYNCcnt path).
// ---------------------------------------------------------------------------
__global__ __launch_bounds__(256) void frame_attn(
    const unsigned short* __restrict__ qws, const unsigned short* __restrict__ kws,
    const unsigned short* __restrict__ vws, unsigned short* __restrict__ ows)
{
    __shared__ __align__(16) unsigned short sK[197 * 64];   // [key][d]
    __shared__ __align__(16) unsigned short sV[64 * 224];   // [d][key] transposed
    __shared__ __align__(16) unsigned short sP[8][16 * 32]; // per-wave P staging

    const int bh = blockIdx.x >> 3;
    const int fr = blockIdx.x & 7;
    const size_t base = (size_t)bh * NTOK * DHEAD;
    const int tid = threadIdx.x, lane = tid & 31, wv = tid >> 5;

#if HAVE_ASYNC_LDS
    // K tile: async DMA rows into LDS, 16B per lane, 197*8 chunks
    for (int i = tid; i < 197 * 8; i += 256) {
        int s = i >> 3, c8 = (i & 7) << 3;
        int gn = (s == 0) ? 0 : (1 + fr * NFR + (s - 1));
        async_ld_b128(kws + base + (size_t)gn * DHEAD + c8, &sK[s * 64 + c8]);
    }
#endif
    // V tile: transposed scatter into LDS (manual); K fallback if no async
    for (int s = tid; s < 224; s += 256) {
        int gn = (s == 0) ? 0 : ((s <= NFR) ? (1 + fr * NFR + (s - 1)) : -1);
#if !HAVE_ASYNC_LDS
        if (s < 197) {
            const ushort8* src = (const ushort8*)(kws + base + (size_t)gn * DHEAD);
            ushort8* dstrow = (ushort8*)(&sK[s * 64]);
            for (int c = 0; c < 8; ++c) dstrow[c] = src[c];
        }
#endif
        if (gn >= 0) {
            const unsigned short* vs = vws + base + (size_t)gn * DHEAD;
            for (int d = 0; d < 64; ++d) sV[d * 224 + s] = vs[d];
        } else {
            for (int d = 0; d < 64; ++d) sV[d * 224 + s] = 0;
        }
    }
#if HAVE_ASYNC_LDS
    __builtin_amdgcn_s_wait_asynccnt(0);
#endif
    __syncthreads();

    const int arow = lane & 15;
    const int kb   = (lane >> 4) * 8;
    const int kh16 = (lane >> 4) * 16;

    for (int qt = wv; qt < 13; qt += 8) {
        int q0 = qt * 16;
        int qrow = q0 + arow; if (qrow > NFR - 1) qrow = NFR - 1;  // clamp pad rows
        const unsigned short* qp = qws + base + (size_t)(1 + fr * NFR + qrow) * DHEAD;
        Frag qf[2];
        qf[0].h[0] = *(const ushort8*)(qp + kh16);
        qf[0].h[1] = *(const ushort8*)(qp + kh16 + 8);
        qf[1].h[0] = *(const ushort8*)(qp + 32 + kh16);
        qf[1].h[1] = *(const ushort8*)(qp + 32 + kh16 + 8);

        floatx8 O[4] = {};
        float mrow[8], lrow[8];
        for (int i = 0; i < 8; ++i) { mrow[i] = -1e30f; lrow[i] = 0.f; }

        for (int st = 0; st < 7; ++st) {          // 7 x 32 = 224 padded keys
            floatx8 s0 = {}, s1 = {};
            int kr0 = st * 32 + (lane & 15);      if (kr0 > 196) kr0 = 196;
            int kr1 = st * 32 + 16 + (lane & 15); if (kr1 > 196) kr1 = 196;
            const unsigned short* kp0 = &sK[kr0 * 64];
            const unsigned short* kp1 = &sK[kr1 * 64];
            Frag kf;
            kf.h[0] = *(const ushort8*)(kp0 + kh16);
            kf.h[1] = *(const ushort8*)(kp0 + kh16 + 8);
            s0 = wmma_bf16(qf[0], kf, s0);
            kf.h[0] = *(const ushort8*)(kp0 + 32 + kh16);
            kf.h[1] = *(const ushort8*)(kp0 + 32 + kh16 + 8);
            s0 = wmma_bf16(qf[1], kf, s0);
            kf.h[0] = *(const ushort8*)(kp1 + kh16);
            kf.h[1] = *(const ushort8*)(kp1 + kh16 + 8);
            s1 = wmma_bf16(qf[0], kf, s1);
            kf.h[0] = *(const ushort8*)(kp1 + 32 + kh16);
            kf.h[1] = *(const ushort8*)(kp1 + 32 + kh16 + 8);
            s1 = wmma_bf16(qf[1], kf, s1);

            // mask padded key columns
            int c0 = st * 32 + (lane & 15);
            int c1 = c0 + 16;
            if (c0 >= 197) for (int i = 0; i < 8; ++i) s0[i] = -1e30f;
            if (c1 >= 197) for (int i = 0; i < 8; ++i) s1[i] = -1e30f;

            // online softmax: row reductions across 16-lane halves
            float al[8];
            for (int i = 0; i < 8; ++i) {
                float c = fmaxf(s0[i], s1[i]);
                for (int off = 1; off < 16; off <<= 1)
                    c = fmaxf(c, __shfl_xor(c, off, 32));
                float nm = fmaxf(mrow[i], c);
                al[i] = exp2f((mrow[i] - nm) * L2E);
                mrow[i] = nm;
                lrow[i] *= al[i];
            }
            for (int nf = 0; nf < 4; ++nf)
                for (int i = 0; i < 8; ++i)
                    O[nf][i] *= al[i];

            float p0[8], p1[8];
            for (int i = 0; i < 8; ++i) {
                p0[i] = exp2f((s0[i] - mrow[i]) * L2E);
                p1[i] = exp2f((s1[i] - mrow[i]) * L2E);
                float r = p0[i] + p1[i];
                for (int off = 1; off < 16; off <<= 1)
                    r += __shfl_xor(r, off, 32);
                lrow[i] += r;
            }

            // transpose P (C layout -> A layout) through per-wave LDS
            unsigned short* pw = sP[wv];
            int prb = (lane >> 4) * 8, pc = lane & 15;
            for (int i = 0; i < 8; ++i) {
                pw[(prb + i) * 32 + pc]      = f2bf(p0[i]);
                pw[(prb + i) * 32 + pc + 16] = f2bf(p1[i]);
            }
            Frag pf;
            const unsigned short* pp = &pw[(lane & 15) * 32];
            pf.h[0] = *(const ushort8*)(pp + kh16);
            pf.h[1] = *(const ushort8*)(pp + kh16 + 8);

            for (int nf = 0; nf < 4; ++nf) {
                Frag vf;
                const unsigned short* vp = &sV[(nf * 16 + (lane & 15)) * 224 + st * 32 + kh16];
                vf.h[0] = *(const ushort8*)(vp);
                vf.h[1] = *(const ushort8*)(vp + 8);
                O[nf] = wmma_bf16(pf, vf, O[nf]);
            }
        }

        for (int nf = 0; nf < 4; ++nf)
            for (int i = 0; i < 8; ++i) {
                int qr = q0 + (lane >> 4) * 8 + i;
                if (qr < NFR) {
                    size_t idx = base + (size_t)(1 + fr * NFR + qr) * DHEAD
                               + nf * 16 + (lane & 15);
                    ows[idx] = f2bf(O[nf][i] / lrow[i]);
                }
            }
    }
}

// ---------------------------------------------------------------------------
// Kernel 4: out = attn_out @ w_proj + b_proj  (bf16 head-major A -> fp32 out)
// Same skeleton as gemm_qkv, K-step 64.
// ---------------------------------------------------------------------------
__global__ __launch_bounds__(256) void gemm_proj(
    const unsigned short* __restrict__ ows, const float* __restrict__ w,
    const float* __restrict__ bias, float* __restrict__ out)
{
    __shared__ __align__(16) unsigned short sA[128 * 72];
    __shared__ __align__(16) unsigned short sB[64 * 72];

    const int m0   = blockIdx.x * 128;
    const int n0   = blockIdx.y * 64;
    const int tid  = threadIdx.x;
    const int lane = tid & 31;
    const int wv   = tid >> 5;
    const int mw   = (wv & 3) * 32;
    const int nw   = (wv >> 2) * 32;
    const int arow = lane & 15;
    const int kb   = (lane >> 4) * 8;
    const int kh16 = (lane >> 4) * 16;

    floatx8 acc[2][2] = {};

    for (int k0 = 0; k0 < DMODEL; k0 += 64) {
        // A tile 128x64 bf16 from head-major attention output (1024 chunks of 8)
        for (int id = tid; id < 1024; id += 256) {
            int row = id >> 3, c8 = (id & 7) << 3;
            int gm = m0 + row;
            ushort8 hval = {0, 0, 0, 0, 0, 0, 0, 0};
            if (gm < MROWS) {
                int b = gm / NTOK, n = gm - b * NTOK;
                int kk = k0 + c8;
                int head = kk >> 6, dc = kk & 63;  // 8-elem run stays inside a head
                hval = *(const ushort8*)(ows +
                        (((size_t)b * NHEAD + head) * NTOK + n) * DHEAD + dc);
            }
            *(ushort8*)(&sA[row * 72 + c8]) = hval;
        }
        // B tile 64x64 fp32 from w_proj, transposed bf16
        for (int id = tid; id < 1024; id += 256) {
            int kr = id >> 4, n4 = (id & 15) << 2;
            float4 f = *(const float4*)(w + (size_t)(k0 + kr) * DMODEL + n0 + n4);
            sB[(n4 + 0) * 72 + kr] = f2bf(f.x);
            sB[(n4 + 1) * 72 + kr] = f2bf(f.y);
            sB[(n4 + 2) * 72 + kr] = f2bf(f.z);
            sB[(n4 + 3) * 72 + kr] = f2bf(f.w);
        }
        if (k0 + 64 < DMODEL)
            __builtin_prefetch(w + (size_t)(k0 + 64 + (tid & 63)) * DMODEL
                                 + n0 + (tid >> 6) * 16, 0, 0);
        __syncthreads();

        for (int kc = 0; kc < 2; ++kc) {
            Frag af[2], bf_[2];
            for (int mf = 0; mf < 2; ++mf) {
                const unsigned short* p = &sA[(mw + mf * 16 + arow) * 72 + kc * 32];
                af[mf].h[0] = *(const ushort8*)(p + kh16);
                af[mf].h[1] = *(const ushort8*)(p + kh16 + 8);
            }
            for (int nf = 0; nf < 2; ++nf) {
                const unsigned short* p = &sB[(nw + nf * 16 + (lane & 15)) * 72 + kc * 32 + kh16];
                bf_[nf].h[0] = *(const ushort8*)(p);
                bf_[nf].h[1] = *(const ushort8*)(p + 8);
            }
            for (int mf = 0; mf < 2; ++mf)
                for (int nf = 0; nf < 2; ++nf)
                    acc[mf][nf] = wmma_bf16(af[mf], bf_[nf], acc[mf][nf]);
        }
        __syncthreads();
    }

    for (int mf = 0; mf < 2; ++mf)
        for (int nf = 0; nf < 2; ++nf) {
            int gcol = n0 + nw + nf * 16 + (lane & 15);
            float bv = bias[gcol];
            for (int i = 0; i < 8; ++i) {
                int gm = m0 + mw + mf * 16 + (lane >> 4) * 8 + i;
                if (gm < MROWS)
                    out[(size_t)gm * DMODEL + gcol] = acc[mf][nf][i] + bv;
            }
        }
}

// ---------------------------------------------------------------------------
// Launch
// ---------------------------------------------------------------------------
extern "C" void kernel_launch(void* const* d_in, const int* in_sizes, int n_in,
                              void* d_out, int out_size, void* d_ws, size_t ws_size,
                              hipStream_t stream) {
    const float* x      = (const float*)d_in[0];
    const float* w_qkv  = (const float*)d_in[1];
    const float* w_proj = (const float*)d_in[2];
    const float* b_proj = (const float*)d_in[3];
    // d_in[4] = f (== 8), baked into the kernels

    const size_t HEADTENS = (size_t)BATCH * NHEAD * NTOK * DHEAD; // 19,279,872 elems
    unsigned short* q_ws = (unsigned short*)d_ws;
    unsigned short* k_ws = q_ws + HEADTENS;
    unsigned short* v_ws = k_ws + HEADTENS;
    unsigned short* o_ws = v_ws + HEADTENS;
    float* out = (float*)d_out;

    dim3 gqkv((MROWS + 127) / 128, (3 * DMODEL) / 64);   // (197, 36)
    gemm_qkv<<<gqkv, 256, 0, stream>>>(x, w_qkv, q_ws, k_ws, v_ws);

    cls_attn<<<BATCH * NHEAD, 256, 0, stream>>>(q_ws, k_ws, v_ws, o_ws);

    frame_attn<<<BATCH * NHEAD * 8, 256, 0, stream>>>(q_ws, k_ws, v_ws, o_ws);

    dim3 gproj((MROWS + 127) / 128, DMODEL / 64);        // (197, 12)
    gemm_proj<<<gproj, 256, 0, stream>>>(o_ws, w_proj, b_proj, out);
}
